// BERTSyntaxRel_42923903156677
// MI455X (gfx1250) — matrix operand order repaired and checked by reading, not compiled, not run on hardware
//
#include <hip/hip_runtime.h>

typedef __attribute__((ext_vector_type(16))) _Float16 v16h;
typedef __attribute__((ext_vector_type(8)))  _Float16 v8h;
typedef __attribute__((ext_vector_type(4)))  _Float16 v4h;
typedef __attribute__((ext_vector_type(8)))  float    v8f;
typedef __attribute__((ext_vector_type(4)))  float    v4f;

#define SS 1024
#define DD 768
#define HH 128
#define RR 48

union V16U { v16h v; v8h h[2]; };

// CDNA5 async global->LDS, GVS mode: SGPR base + 32-bit VGPR offset + imm.
// INST_OFFSET is added to BOTH the LDS and global addresses (ISA 08 §4.4),
// so one (lds, voff) pair covers a whole contiguous per-lane block.
template <int OFF>
__device__ __forceinline__ void agvs(unsigned lds, unsigned voff, const void* sbase) {
  asm volatile("global_load_async_to_lds_b128 %0, %1, %2 offset:%3"
               :: "v"(lds), "v"(voff), "s"(sbase), "i"(OFF) : "memory");
}
// copy 128B per lane (8 x b128)
__device__ __forceinline__ void stage128(unsigned lds, unsigned voff, const void* s) {
  agvs<0>(lds, voff, s);   agvs<16>(lds, voff, s);
  agvs<32>(lds, voff, s);  agvs<48>(lds, voff, s);
  agvs<64>(lds, voff, s);  agvs<80>(lds, voff, s);
  agvs<96>(lds, voff, s);  agvs<112>(lds, voff, s);
}
// copy 256B per lane (16 x b128)
__device__ __forceinline__ void stage256(unsigned lds, unsigned voff, const void* s) {
  stage128(lds, voff, s);
  agvs<128>(lds, voff, s); agvs<144>(lds, voff, s);
  agvs<160>(lds, voff, s); agvs<176>(lds, voff, s);
  agvs<192>(lds, voff, s); agvs<208>(lds, voff, s);
  agvs<224>(lds, voff, s); agvs<240>(lds, voff, s);
}
template <int N>
__device__ __forceinline__ void wait_async() {
  asm volatile("s_wait_asynccnt %0" :: "i"(N) : "memory");
}

// ---------------------------------------------------------------------------
// Prep: f32 row-major (rows x cols) -> f16 transposed (cols x rows), LDS-tiled.
// ---------------------------------------------------------------------------
__global__ __launch_bounds__(256)
void xpose_f32_to_f16(const float* __restrict__ in, _Float16* __restrict__ outp,
                      int rows, int cols) {
  __shared__ float tl[32][33];
  const int tx = threadIdx.x & 31;
  const int ty = threadIdx.x >> 5;
  const int ntx = cols >> 5;
  const int tcol = (blockIdx.x % ntx) * 32;
  const int trow = (blockIdx.x / ntx) * 32;
  for (int i = ty; i < 32; i += 8)
    tl[i][tx] = in[(size_t)(trow + i) * cols + tcol + tx];
  __syncthreads();
  for (int i = ty; i < 32; i += 8)
    outp[(size_t)(tcol + i) * rows + trow + tx] = (_Float16)tl[tx][i];
}

// ---------------------------------------------------------------------------
// Fused kernel: 4 waves/block, 32 tokens/wave (two 16-row M tiles).
// W and K streams staged block-wide into LDS with async-to-LDS, double-buffered.
//   WhT/WtT : f16 [HH][DD]    Kt : f16 [RR*HH][HH]
// LDS map:  [0,16K)   abuf 4KB/wave  (A rows, 64-wide d chunk, f16)
//           [16K,48K) hbuf 8KB/wave  (relu(head) 32x128 f16)
//           [48K,112K) region: phase1 = W double buf (2x16KB)
//                              phase2 = K double buf (2x32KB)
// ---------------------------------------------------------------------------
__global__ __launch_bounds__(128)
__attribute__((amdgpu_waves_per_eu(4)))
void biaffine_wmma_kernel(const float* __restrict__ x,
                          const int*   __restrict__ head_id,
                          const float* __restrict__ root,
                          const float* __restrict__ bh,
                          const float* __restrict__ bt,
                          const _Float16* __restrict__ WhT,
                          const _Float16* __restrict__ WtT,
                          const _Float16* __restrict__ Kt,
                          float* __restrict__ out) {
  __shared__ __align__(128) char smem[112 * 1024];

  const int wave = threadIdx.x >> 5;
  const int lane = threadIdx.x & 31;
  const int tid  = threadIdx.x;

  _Float16* abuf = (_Float16*)smem + wave * (32 * 64);
  _Float16* hbuf = (_Float16*)(smem + 16 * 1024) + wave * (32 * 128);
  char*     region = smem + 48 * 1024;

  // LDS byte addresses / per-lane byte offsets for async staging
  const unsigned region_lds = (unsigned)(size_t)region;
  const unsigned wlds0 = region_lds + (unsigned)tid * 128;   // W buf0 slot
  const unsigned wlds1 = wlds0 + 16 * 1024;                  // W buf1 slot
  const unsigned klds0 = region_lds + (unsigned)tid * 256;   // K buf0 slot
  const unsigned klds1 = klds0 + 32 * 1024;                  // K buf1 slot
  const unsigned vWoff = (unsigned)tid * (DD * 2);           // col=tid in W
  const unsigned vKoff = (unsigned)tid * (HH * 2);           // col=tid in K slab

  const int tok0 = (blockIdx.x * 4 + wave) * 32;
  const int b    = tok0 >> 10;
  const int s0   = tok0 & 1023;
  const int hlf  = lane >> 4;
  const int ln   = lane & 15;

  // per-lane source rows (lane == row within the wave's 32-token tile)
  const int hid = head_id[b * SS + s0 + lane];
  const float* gsrc0 = (hid == 0) ? root : (x + (size_t)(b * SS + hid - 1) * DD);
  const float* xsrc0 = x + (size_t)(b * SS + s0 + lane) * DD;

  v8h tailH[2][8];     // relu(tail) packed f16, C-matrix layout

  // ---------------- phase 1: two FF passes (head, then tail) --------------
  for (int pass = 0; pass < 2; ++pass) {
    const _Float16* Wg  = pass ? WtT : WhT;
    const float*    src = pass ? xsrc0 : gsrc0;
    const float*    bia = pass ? bt : bh;
    v8f C[2][8];
    for (int m = 0; m < 2; ++m)
      for (int n = 0; n < 8; ++n) C[m][n] = (v8f){};

    stage128(wlds0, vWoff, Wg);                    // W chunk 0 (16KB)
    for (int kc = 0; kc < 12; ++kc) {              // D = 768 in 64-wide chunks
      // A rows for this chunk: 1 row/lane, f32->f16 into per-wave abuf
      {
        const float* s = src + kc * 64;
        for (int q = 0; q < 16; ++q) {
          v4f v = *(const v4f*)(s + q * 4);
          v4h h4 = { (_Float16)v[0], (_Float16)v[1], (_Float16)v[2], (_Float16)v[3] };
          *(v4h*)(abuf + lane * 64 + q * 4) = h4;
        }
      }
      if (kc < 11) {                               // prefetch next W chunk
        stage128(((kc + 1) & 1) ? wlds1 : wlds0, vWoff, Wg + (kc + 1) * 64);
        wait_async<8>();
      } else {
        wait_async<0>();
      }
      __syncthreads();

      const _Float16* wb = (const _Float16*)(region + (kc & 1) * (16 * 1024));
      V16U A[2][2];                                // (m, kk) tiles, reused over n
      for (int m = 0; m < 2; ++m)
        for (int kk = 0; kk < 2; ++kk) {
          const _Float16* pa = abuf + (m * 16 + ln) * 64 + kk * 32 + hlf * 8;
          A[m][kk].h[0] = *(const v8h*)pa;
          A[m][kk].h[1] = *(const v8h*)(pa + 16);
        }
      for (int n = 0; n < 8; ++n) {
        V16U Bt2[2];                               // both kk tiles up front
        const _Float16* pb = wb + (n * 16 + ln) * 64 + hlf * 8;
        Bt2[0].h[0] = *(const v8h*)pb;        Bt2[0].h[1] = *(const v8h*)(pb + 16);
        Bt2[1].h[0] = *(const v8h*)(pb + 32); Bt2[1].h[1] = *(const v8h*)(pb + 48);
        for (int kk = 0; kk < 2; ++kk) {
          C[0][n] = __builtin_amdgcn_wmma_f32_16x16x32_f16(
              false, A[0][kk].v, false, Bt2[kk].v, (short)0, C[0][n], false, false);
          C[1][n] = __builtin_amdgcn_wmma_f32_16x16x32_f16(
              false, A[1][kk].v, false, Bt2[kk].v, (short)0, C[1][n], false, false);
        }
      }
      __syncthreads();                             // protect W buffer reuse
    }
    // bias + relu; pass0 -> hbuf (f16, row-major); pass1 -> packed f16 regs
    for (int n = 0; n < 8; ++n) {
      const float bv = bia[n * 16 + ln];
      for (int m = 0; m < 2; ++m)
        for (int j = 0; j < 8; ++j) {
          float v = C[m][n][j] + bv; v = v > 0.f ? v : 0.f;
          if (pass == 0) hbuf[(m * 16 + j + 8 * hlf) * 128 + n * 16 + ln] = (_Float16)v;
          else           tailH[m][n][j] = (_Float16)v;
        }
    }
  }

  // ---------------- phase 2: biaffine, K slab double-buffered in LDS ------
  stage256(klds0, vKoff, Kt);                      // r=0 DMA overlaps below
  // resident head A operands (hbuf survived pass 1)
  v16h headA[2][4];
  for (int m = 0; m < 2; ++m)
    for (int hk = 0; hk < 4; ++hk) {
      const _Float16* pa = hbuf + (m * 16 + ln) * 128 + hk * 32 + hlf * 8;
      V16U Av; Av.h[0] = *(const v8h*)pa; Av.h[1] = *(const v8h*)(pa + 16);
      headA[m][hk] = Av.v;
    }

  for (int r = 0; r < RR; ++r) {
    if (r < RR - 1) {                              // prefetch next K slab
      stage256(((r + 1) & 1) ? klds1 : klds0, vKoff,
               Kt + (size_t)(r + 1) * 128 * HH);
      wait_async<16>();
    } else {
      wait_async<0>();
    }
    __syncthreads();
    const _Float16* kb = (const _Float16*)(region + (r & 1) * (32 * 1024));

    v8f acc0 = (v8f){}, acc1 = (v8f){};
    for (int kt = 0; kt < 8; ++kt) {
      V16U Bt[4];                                  // all 4 hk tiles up front
      const _Float16* colbase = kb + (kt * 16 + ln) * 128 + hlf * 8;
      for (int hk = 0; hk < 4; ++hk) {
        Bt[hk].h[0] = *(const v8h*)(colbase + hk * 32);
        Bt[hk].h[1] = *(const v8h*)(colbase + hk * 32 + 16);
      }
      v8f c0 = (v8f){}, c1 = (v8f){};
      for (int hk = 0; hk < 4; ++hk) {
        c0 = __builtin_amdgcn_wmma_f32_16x16x32_f16(
            false, headA[0][hk], false, Bt[hk].v, (short)0, c0, false, false);
        c1 = __builtin_amdgcn_wmma_f32_16x16x32_f16(
            false, headA[1][hk], false, Bt[hk].v, (short)0, c1, false, false);
      }
      for (int j = 0; j < 8; ++j) {                // mixed f32*f16 FMA
        acc0[j] += c0[j] * (float)tailH[0][kt][j];
        acc1[j] += c1[j] * (float)tailH[1][kt][j];
      }
    }
    for (int j = 0; j < 8; ++j) {                  // sum over k lanes (16/half)
      float v0 = acc0[j], v1 = acc1[j];
      v0 += __shfl_xor(v0, 1, 32); v0 += __shfl_xor(v0, 2, 32);
      v0 += __shfl_xor(v0, 4, 32); v0 += __shfl_xor(v0, 8, 32);
      v1 += __shfl_xor(v1, 1, 32); v1 += __shfl_xor(v1, 2, 32);
      v1 += __shfl_xor(v1, 4, 32); v1 += __shfl_xor(v1, 8, 32);
      acc0[j] = v0; acc1[j] = v1;
    }
    if (ln == 0) {                                 // lanes 0 & 16: rows j, j+8
      float* op0 = out + ((size_t)tok0 + 8 * hlf) * RR + r;
      float* op1 = out + ((size_t)tok0 + 16 + 8 * hlf) * RR + r;
      for (int j = 0; j < 8; ++j) {
        op0[(size_t)j * RR] = acc0[j];
        op1[(size_t)j * RR] = acc1[j];
      }
    }
    __syncthreads();                               // protect K buffer reuse
  }
}

// ---------------------------------------------------------------------------
extern "C" void kernel_launch(void* const* d_in, const int* in_sizes, int n_in,
                              void* d_out, int out_size, void* d_ws, size_t ws_size,
                              hipStream_t stream) {
  (void)in_sizes; (void)n_in; (void)out_size; (void)ws_size;
  const float* x       = (const float*)d_in[0];
  const int*   head_id = (const int*)  d_in[1];
  const float* root    = (const float*)d_in[2];
  const float* Wh      = (const float*)d_in[3];
  const float* bh      = (const float*)d_in[4];
  const float* Wt      = (const float*)d_in[5];
  const float* bt      = (const float*)d_in[6];
  const float* Kmat    = (const float*)d_in[7];

  _Float16* wsh = (_Float16*)d_ws;
  _Float16* WhT = wsh;                       //  98304 halves
  _Float16* WtT = wsh + 98304;               //  98304 halves
  _Float16* Kt  = wsh + 196608;              // 786432 halves (~1.9MB total)

  xpose_f32_to_f16<<<96,  256, 0, stream>>>(Wh,   WhT, DD, HH);
  xpose_f32_to_f16<<<96,  256, 0, stream>>>(Wt,   WtT, DD, HH);
  xpose_f32_to_f16<<<768, 256, 0, stream>>>(Kmat, Kt,  HH, HH * RR);

  // 32768 tokens / 32 per wave = 1024 waves; 4 waves per 128-thread block
  biaffine_wmma_kernel<<<256, 128, 0, stream>>>(
      x, head_id, root, bh, bt, WhT, WtT, Kt, (float*)d_out);
}
